// GPRGNNV2_9345848836282
// MI455X (gfx1250) — compile-verified, matrix operand order checked by reading, not run on hardware
//
#include <hip/hip_runtime.h>

// ---------------------------------------------------------------------------
// GPRGNN on MI455X (gfx1250, wave32)
//   x      = relu(feature @ W1 + b1)            [N,H]   -- WMMA bf16x3 GEMM
//   hidden = gamma0 * x                                  (fused in epilogue)
//   repeat K: xnew[dst] += norm*x[src] (atomic scatter); hidden += gamma_k*xnew
//   out    = hidden @ W2 + b2                   [N,C]   -- WMMA bf16x3 GEMM
// ---------------------------------------------------------------------------

typedef __attribute__((ext_vector_type(16))) __bf16 v16bf;
typedef __attribute__((ext_vector_type(8)))  float  v8f;

namespace {
constexpr int kN  = 50000;   // nodes (3125 row-tiles of 16)
constexpr int kIN = 500;     // in_feats
constexpr int kH  = 128;     // hidden (8 col-tiles of 16)
constexpr int kC  = 40;      // classes (3 col-tiles of 16, last partial)
constexpr int kK  = 10;      // propagation steps; temp is [H, K+1]
constexpr int kTempStride = kK + 1;  // 11
}

// K-index map for 16-bit A (16x32) / B (32x16) WMMA fragments (ISA 7.12.2):
// lane L holds M (or N) = L&15; half j of the v16bf maps to
//   j in [0,8):  k = kb + j          (kb = 8 if lane>=16 else 0)
//   j in [8,16): k = kb + 8 + j      (i.e. kb+16 .. kb+23)
__device__ __forceinline__ int wmma_kmap(int lane, int j) {
    int kb = (lane & 16) ? 8 : 0;
    return (j < 8) ? (kb + j) : (kb + 8 + j);
}

// Native device-scope f32 atomic add (CDNA5 GLOBAL_ATOMIC_ADD_F32, no return).
__device__ __forceinline__ void atomic_add_f32_dev(float* p, float v) {
    asm volatile("global_atomic_add_f32 %0, %1, off scope:SCOPE_DEV"
                 :: "v"(p), "v"(v) : "memory");
}

// ---------------------------------------------------------------------------
// GEMM1: x = relu(feature @ W1 + b1); hidden = gamma0 * x
// One wave per 16x16 output tile. 25000 waves total (3125 x 8), exact grid.
// bf16x3 split accumulation for ~fp32 accuracy on the matrix pipes.
// ---------------------------------------------------------------------------
__global__ void __launch_bounds__(128)
gemm1_relu_kernel(const float* __restrict__ A,     // [N, IN]
                  const float* __restrict__ B,     // [IN, H]
                  const float* __restrict__ bias,  // [H]
                  const float* __restrict__ temp,  // [H, K+1]
                  float* __restrict__ X,           // [N, H]
                  float* __restrict__ Hd) {        // [N, H]
    const int lane = threadIdx.x & 31;
    const int wid  = (blockIdx.x * blockDim.x + threadIdx.x) >> 5;
    const int mt   = wid >> 3;             // 0..3124
    const int nt   = wid & 7;              // 0..7
    const int row  = mt * 16 + (lane & 15);
    const int bcol = nt * 16 + (lane & 15);

    v8f acc = {};
    for (int kc = 0; kc < kIN; kc += 32) {
        float af[16], bf[16];
#pragma unroll
        for (int j = 0; j < 16; ++j) {
            const int k = kc + wmma_kmap(lane, j);
            const bool ok = (k < kIN);     // pad K=500 -> 512 with zeros
            af[j] = ok ? A[(size_t)row * kIN + k] : 0.0f;
            bf[j] = ok ? B[(size_t)k * kH + bcol] : 0.0f;
        }
        v16bf ahi, alo, bhi, blo;
#pragma unroll
        for (int j = 0; j < 16; ++j) {
            __bf16 h = (__bf16)af[j];
            ahi[j] = h;  alo[j] = (__bf16)(af[j] - (float)h);
            __bf16 g = (__bf16)bf[j];
            bhi[j] = g;  blo[j] = (__bf16)(bf[j] - (float)g);
        }
        acc = __builtin_amdgcn_wmma_f32_16x16x32_bf16(false, ahi, false, bhi,
                                                      (short)0, acc, false, false);
        acc = __builtin_amdgcn_wmma_f32_16x16x32_bf16(false, ahi, false, blo,
                                                      (short)0, acc, false, false);
        acc = __builtin_amdgcn_wmma_f32_16x16x32_bf16(false, alo, false, bhi,
                                                      (short)0, acc, false, false);
    }

    const float bv = bias[bcol];
    const float g0 = temp[bcol * kTempStride + 0];
#pragma unroll
    for (int r = 0; r < 8; ++r) {
        const int  orow = mt * 16 + ((lane < 16) ? r : r + 8);
        float v = acc[r] + bv;
        v = v > 0.0f ? v : 0.0f;           // relu
        const size_t off = (size_t)orow * kH + bcol;
        X[off]  = v;
        Hd[off] = v * g0;
    }
}

// ---------------------------------------------------------------------------
// Edge scatter: one wave per edge; lane l handles channels [4l, 4l+4).
// 512B gather of x[src] (L2-resident) + 128 native f32 atomics into xnew[dst].
// ---------------------------------------------------------------------------
__global__ void __launch_bounds__(256)
edge_kernel(const float* __restrict__ x, float* __restrict__ xnew,
            const int* __restrict__ src, const int* __restrict__ dst,
            const float* __restrict__ norm, int E) {
    const int e    = (blockIdx.x * blockDim.x + threadIdx.x) >> 5;
    const int lane = threadIdx.x & 31;
    if (e >= E) return;
    const int   s = src[e];
    const int   d = dst[e];
    const float w = norm[e];
    const float4 v = ((const float4*)(x + (size_t)s * kH))[lane];
    float* p = xnew + (size_t)d * kH + lane * 4;
    atomic_add_f32_dev(p + 0, w * v.x);
    atomic_add_f32_dev(p + 1, w * v.y);
    atomic_add_f32_dev(p + 2, w * v.z);
    atomic_add_f32_dev(p + 3, w * v.w);
}

// ---------------------------------------------------------------------------
// hidden += gamma_k * xnew, and zero the retired buffer so it can serve as
// the next iteration's accumulator (ping-pong). Exact grid: NH/4/256 = 6250.
// ---------------------------------------------------------------------------
__global__ void __launch_bounds__(256)
update_kernel(float4* __restrict__ hidden, const float4* __restrict__ xnew,
              float4* __restrict__ xzero, const float* __restrict__ temp,
              int kidx) {
    const size_t i = (size_t)blockIdx.x * 256 + threadIdx.x;  // over N*H/4
    const int h = (int)(i & (kH / 4 - 1)) * 4;                // channel base
    float4 xv = xnew[i];
    float4 hv = hidden[i];
    hv.x += temp[(h + 0) * kTempStride + kidx] * xv.x;
    hv.y += temp[(h + 1) * kTempStride + kidx] * xv.y;
    hv.z += temp[(h + 2) * kTempStride + kidx] * xv.z;
    hv.w += temp[(h + 3) * kTempStride + kidx] * xv.w;
    hidden[i] = hv;
    xzero[i] = make_float4(0.f, 0.f, 0.f, 0.f);
}

__global__ void __launch_bounds__(256)
zero_kernel(float4* __restrict__ p) {
    const size_t i = (size_t)blockIdx.x * 256 + threadIdx.x;
    p[i] = make_float4(0.f, 0.f, 0.f, 0.f);
}

// ---------------------------------------------------------------------------
// GEMM2: out = hidden @ W2 + b2.  [50000x128]@[128x40]; pad C=40 -> 48 cols.
// 9375 waves (3125 x 3); wave-uniform guard keeps EXEC all-ones for WMMA.
// ---------------------------------------------------------------------------
__global__ void __launch_bounds__(128)
gemm2_kernel(const float* __restrict__ A,     // [N, H]
             const float* __restrict__ B,     // [H, C]
             const float* __restrict__ bias,  // [C]
             float* __restrict__ Out) {       // [N, C]
    const int lane = threadIdx.x & 31;
    const int wid  = (blockIdx.x * blockDim.x + threadIdx.x) >> 5;
    if (wid >= (kN / 16) * 3) return;          // wave-uniform
    const int mt   = wid / 3;
    const int nt   = wid % 3;
    const int row  = mt * 16 + (lane & 15);
    const int bcol = nt * 16 + (lane & 15);

    v8f acc = {};
    for (int kc = 0; kc < kH; kc += 32) {
        float af[16], bf[16];
#pragma unroll
        for (int j = 0; j < 16; ++j) {
            const int k = kc + wmma_kmap(lane, j);
            af[j] = A[(size_t)row * kH + k];
            bf[j] = (bcol < kC) ? B[(size_t)k * kC + bcol] : 0.0f;
        }
        v16bf ahi, alo, bhi, blo;
#pragma unroll
        for (int j = 0; j < 16; ++j) {
            __bf16 h = (__bf16)af[j];
            ahi[j] = h;  alo[j] = (__bf16)(af[j] - (float)h);
            __bf16 g = (__bf16)bf[j];
            bhi[j] = g;  blo[j] = (__bf16)(bf[j] - (float)g);
        }
        acc = __builtin_amdgcn_wmma_f32_16x16x32_bf16(false, ahi, false, bhi,
                                                      (short)0, acc, false, false);
        acc = __builtin_amdgcn_wmma_f32_16x16x32_bf16(false, ahi, false, blo,
                                                      (short)0, acc, false, false);
        acc = __builtin_amdgcn_wmma_f32_16x16x32_bf16(false, alo, false, bhi,
                                                      (short)0, acc, false, false);
    }

    if (bcol < kC) {                           // per-lane store guard is fine
        const float bv = bias[bcol];
#pragma unroll
        for (int r = 0; r < 8; ++r) {
            const int orow = mt * 16 + ((lane < 16) ? r : r + 8);
            Out[(size_t)orow * kC + bcol] = acc[r] + bv;
        }
    }
}

// ---------------------------------------------------------------------------
extern "C" void kernel_launch(void* const* d_in, const int* in_sizes, int n_in,
                              void* d_out, int out_size, void* d_ws, size_t ws_size,
                              hipStream_t stream) {
    const float* feature    = (const float*)d_in[0];  // [N, IN]
    const float* W1         = (const float*)d_in[1];  // [IN, H]
    const float* b1         = (const float*)d_in[2];  // [H]
    const float* W2         = (const float*)d_in[3];  // [H, C]
    const float* b2         = (const float*)d_in[4];  // [C]
    const float* temp       = (const float*)d_in[5];  // [H, K+1]
    const float* norm       = (const float*)d_in[6];  // [E]
    const int*   edge_index = (const int*)d_in[7];    // [2, E] flat

    const int E = in_sizes[6];
    const int* src = edge_index;
    const int* dst = edge_index + E;

    const size_t NH      = (size_t)kN * kH;           // 6.4M floats
    const size_t NHbytes = NH * sizeof(float);        // 25.6 MB
    char* ws = (char*)d_ws;
    float* bufA = (float*)(ws);                       // x       (ping)
    float* bufB = (float*)(ws + NHbytes);             // xnew    (pong)
    float* bufH = (float*)(ws + 2 * NHbytes);         // hidden

    const int elemBlocks = (int)(NH / 4 / 256);       // 6250, exact
    const int gemm1Blocks = (kN / 16) * (kH / 16) / 4;          // 6250 (4 waves/blk)
    const int gemm2Blocks = ((kN / 16) * 3 + 3) / 4;            // 2344
    const int edgeBlocks  = (E + 7) / 8;                        // 8 waves/blk

    // xnew must start at zero (buffers are poisoned, and accumulated into).
    zero_kernel<<<elemBlocks, 256, 0, stream>>>((float4*)bufB);

    // x = relu(feature@W1+b1); hidden = gamma0*x  (fused epilogue)
    gemm1_relu_kernel<<<gemm1Blocks, 128, 0, stream>>>(feature, W1, b1, temp,
                                                       bufA, bufH);

    float* cur = bufA;
    float* nxt = bufB;
    for (int k = 0; k < kK; ++k) {
        edge_kernel<<<edgeBlocks, 256, 0, stream>>>(cur, nxt, src, dst, norm, E);
        // hidden += gamma_{k+1} * xnew; zero retired buffer for next round
        update_kernel<<<elemBlocks, 256, 0, stream>>>((float4*)bufH,
                                                      (const float4*)nxt,
                                                      (float4*)cur, temp, k + 1);
        float* t = cur; cur = nxt; nxt = t;
    }

    // out = hidden @ W2 + b2
    gemm2_kernel<<<gemm2Blocks, 128, 0, stream>>>(bufH, W2, b2, (float*)d_out);
}
